// GroupQueryAttention_78898549228111
// MI455X (gfx1250) — compile-verified, matrix operand order checked by reading, not run on hardware
//
#include <hip/hip_runtime.h>
#include <stdint.h>

#define EMB  2048
#define HD   64
#define NH   32
#define NKV  8
#define GS   4
#define WIN  128
#define BATCH 2
#define SEQ  2048
#define TOK  (BATCH*SEQ)            // 4096 tokens
#define NQKV (NH*HD + 2*NKV*HD)     // 3072 = 2048 Q | 512 K | 512 V

#ifndef __has_builtin
#define __has_builtin(x) 0
#endif
#if __has_builtin(__builtin_amdgcn_tensor_load_to_lds) && \
    __has_builtin(__builtin_amdgcn_s_wait_tensorcnt)
#define USE_TDM 1
#else
#define USE_TDM 0
#endif

typedef __attribute__((ext_vector_type(16))) __bf16 v16bf;
typedef __attribute__((ext_vector_type(8)))  float  v8f;
typedef unsigned int v4u  __attribute__((ext_vector_type(4)));
typedef int          v8i_ __attribute__((ext_vector_type(8)));
typedef int          v4i_ __attribute__((ext_vector_type(4)));

struct U4 { unsigned x, y, z, w; };
union BFrag { unsigned u[8]; v16bf v; };

__device__ __forceinline__ unsigned short f2bf(float f) {
  unsigned u = __float_as_uint(f);
  u += 0x7FFFu + ((u >> 16) & 1u);           // round-to-nearest-even
  return (unsigned short)(u >> 16);
}
__device__ __forceinline__ float bf2f(unsigned short h) {
  return __uint_as_float(((unsigned)h) << 16);
}

// 16-bit A/B fragment load per CDNA5 WMMA layout:
// lane half=0 holds K = {kb+0..7, kb+16..23}, half=1 holds K = {kb+8..15, kb+24..31}
__device__ __forceinline__ void load_frag(BFrag& f, const unsigned short* rowptr,
                                          int kb, int half) {
  const unsigned short* q = rowptr + kb + half * 8;
  *(U4*)(&f.u[0]) = *(const U4*)(q);
  *(U4*)(&f.u[4]) = *(const U4*)(q + 16);
}

__device__ __forceinline__ v8f wmma_bf16(const BFrag& a, const BFrag& b, v8f c) {
  return __builtin_amdgcn_wmma_f32_16x16x32_bf16(false, a.v, false, b.v,
                                                 (short)0, c, false, false);
}

#if USE_TDM
// Issue a 2D TDM tile load: tile_d1 rows of tile_d0 bf16 elements, row stride
// stride0 elements, into LDS byte offset lds_off (rows packed contiguously).
__device__ __forceinline__ void tdm_load_2d(unsigned lds_off, const void* gptr,
                                            unsigned tile_d0, unsigned tile_d1,
                                            unsigned long long stride0) {
  unsigned long long ga = (unsigned long long)gptr;
  v4u g0;
  g0[0] = 1u;                                        // count=1, user descriptor
  g0[1] = lds_off;                                   // lds_addr
  g0[2] = (unsigned)(ga & 0xFFFFFFFFu);              // global_addr[31:0]
  g0[3] = ((unsigned)(ga >> 32) & 0x01FFFFFFu) | (2u << 30);  // addr[56:32]|type=2
  v8i_ g1;
  g1[0] = (int)(1u << 16);                           // data_size=1 (2 bytes)
  g1[1] = (int)((tile_d0 & 0xFFFFu) << 16);          // tensor_dim0[15:0]
  g1[2] = (int)((tile_d0 >> 16) | ((tile_d1 & 0xFFFFu) << 16));  // td0_hi|td1_lo
  g1[3] = (int)((tile_d1 >> 16) | (tile_d0 << 16));  // td1_hi | tile_dim0
  g1[4] = (int)(tile_d1 & 0xFFFFu);                  // tile_dim1, tile_dim2=0
  g1[5] = (int)(unsigned)(stride0 & 0xFFFFFFFFull);  // tensor_dim0_stride[31:0]
  g1[6] = (int)((unsigned)(stride0 >> 32) & 0xFFFFu);// stride0[47:32]
  g1[7] = 0;
  v4i_ z4 = {0, 0, 0, 0};
#if defined(__clang_major__) && __clang_major__ >= 23
  v8i_ z8 = {0, 0, 0, 0, 0, 0, 0, 0};
  __builtin_amdgcn_tensor_load_to_lds(g0, g1, z4, z4, z8, 0);
#else
  __builtin_amdgcn_tensor_load_to_lds(g0, g1, z4, z4, 0);
#endif
}
#endif

// ---------------- fp32 -> bf16 pack ----------------
__global__ void k_f32_to_bf16(const float* __restrict__ src,
                              unsigned short* __restrict__ dst, int n) {
  int i = blockIdx.x * blockDim.x + threadIdx.x;
  int stride = gridDim.x * blockDim.x;
  for (; i < n; i += stride) dst[i] = f2bf(src[i]);
}

// ---------------- GEMM core: 64x64 tile per wave (4x4 register blocking) -----
// 8 fragment loads feed 16 WMMAs per 32-wide k-step (~32 FLOP/byte from L2).
__device__ __forceinline__ void gemm_tile64(const unsigned short* __restrict__ A,
                                            const unsigned short* __restrict__ B,
                                            int row0, int col0, int K,
                                            int idx, int half, v8f acc[4][4]) {
  const unsigned short* Arow[4];
  const unsigned short* Bcol[4];
  for (int t = 0; t < 4; ++t) {
    Arow[t] = A + (size_t)(row0 + t * 16 + idx) * K;
    Bcol[t] = B + (size_t)(col0 + t * 16 + idx) * K;
  }
  for (int kb = 0; kb < K; kb += 32) {
    __builtin_prefetch(Arow[0] + kb + 512, 0, 1);    // global_prefetch_b8
    __builtin_prefetch(Bcol[0] + kb + 512, 0, 1);
    BFrag a[4], b[4];
    for (int t = 0; t < 4; ++t) load_frag(a[t], Arow[t], kb, half);
    for (int t = 0; t < 4; ++t) load_frag(b[t], Bcol[t], kb, half);
    for (int mt = 0; mt < 4; ++mt)
      for (int nt = 0; nt < 4; ++nt)
        acc[mt][nt] = wmma_bf16(a[mt], b[nt], acc[mt][nt]);
  }
}

// QKV projection: [TOK x EMB] x [NQKV x EMB]^T + bias -> bf16 [TOK x NQKV]
__global__ __launch_bounds__(128) void k_gemm_qkv(
    const unsigned short* __restrict__ Xb,
    const unsigned short* __restrict__ Wb,
    const float* __restrict__ bias,
    unsigned short* __restrict__ Out) {
  const int lane = threadIdx.x & 31, idx = lane & 15, half = lane >> 4;
  const int wave = threadIdx.x >> 5;
  const int row0 = blockIdx.y * 64;
  const int col0 = (blockIdx.x * 4 + wave) * 64;
  v8f acc[4][4];
  for (int mt = 0; mt < 4; ++mt) for (int nt = 0; nt < 4; ++nt) acc[mt][nt] = (v8f){};
  gemm_tile64(Xb, Wb, row0, col0, EMB, idx, half, acc);
  for (int nt = 0; nt < 4; ++nt) {
    float b = bias[col0 + nt * 16 + idx];
    for (int mt = 0; mt < 4; ++mt)
      for (int r = 0; r < 8; ++r) {
        int row = row0 + mt * 16 + half * 8 + r;
        Out[(size_t)row * NQKV + col0 + nt * 16 + idx] = f2bf(acc[mt][nt][r] + b);
      }
  }
}

// Output projection: [TOK x EMB] x [EMB x EMB]^T + bo -> fp32 d_out
__global__ __launch_bounds__(128) void k_gemm_out(
    const unsigned short* __restrict__ Cb,
    const unsigned short* __restrict__ Wob,
    const float* __restrict__ bo,
    float* __restrict__ Out) {
  const int lane = threadIdx.x & 31, idx = lane & 15, half = lane >> 4;
  const int wave = threadIdx.x >> 5;
  const int row0 = blockIdx.y * 64;
  const int col0 = (blockIdx.x * 4 + wave) * 64;
  v8f acc[4][4];
  for (int mt = 0; mt < 4; ++mt) for (int nt = 0; nt < 4; ++nt) acc[mt][nt] = (v8f){};
  gemm_tile64(Cb, Wob, row0, col0, EMB, idx, half, acc);
  for (int nt = 0; nt < 4; ++nt) {
    float b = bo[col0 + nt * 16 + idx];
    for (int mt = 0; mt < 4; ++mt)
      for (int r = 0; r < 8; ++r) {
        int row = row0 + mt * 16 + half * 8 + r;
        Out[(size_t)row * EMB + col0 + nt * 16 + idx] = acc[mt][nt][r] + b;
      }
  }
}

// ---------------- YaRN RoPE, in place on Q and K slices ----------------
__global__ void k_rope(unsigned short* __restrict__ QKVb,
                       const int* __restrict__ pos) {
  int i = blockIdx.x * blockDim.x + threadIdx.x;
  const int total = TOK * (NH + NKV) * (HD / 2);
  if (i >= total) return;
  int d  = i & 31;
  int hh = (i >> 5) % (NH + NKV);
  int t  = i / (32 * (NH + NKV));
  int coloff = (hh < NH) ? hh * HD : (NH * HD + (hh - NH) * HD);
  float p   = (float)pos[t];
  float f   = __powf(10000.0f, -(float)d / 32.0f);
  float wl  = 6.28318530717958647f / f;
  float tt  = (wl - 32.0f) * (1.0f / (1024.0f - 32.0f));
  tt = fminf(fmaxf(tt, 0.0f), 1.0f);
  float eff  = f * (1.0f - tt) + 0.25f * f * tt;
  float conc = 0.1f * 1.38629436111989062f + 1.0f;   // 0.1*ln(4)+1
  float ang  = p * eff * conc;
  float c = cosf(ang), s = sinf(ang);
  unsigned short* base = QKVb + (size_t)t * NQKV + coloff;
  float x1 = bf2f(base[d]);
  float x2 = bf2f(base[d + 32]);
  base[d]      = f2bf(x1 * c - x2 * s);
  base[d + 32] = f2bf(x2 * c + x1 * s);
}

// ---------------- V transpose: [token][g*64+d] -> [b][g][d][seq] ----------------
__global__ void k_transpose_v(const unsigned short* __restrict__ QKVb,
                              unsigned short* __restrict__ Vt) {
  int i = blockIdx.x * blockDim.x + threadIdx.x;
  const int total = TOK * NKV * HD;                  // 2^21
  if (i >= total) return;
  int s = i & (SEQ - 1);
  int d = (i >> 11) & (HD - 1);
  int g = (i >> 17) & (NKV - 1);
  int b = i >> 20;
  int t = b * SEQ + s;
  Vt[i] = QKVb[(size_t)t * NQKV + NH * HD + NKV * HD + g * HD + d];
}

// ---------------- sliding-window flash attention with sink ----------------
// Per-wave dynamic LDS region (bytes): K tile 4096 | V tile 4096 | P tile 1024
#define ATTN_LDS_PER_WAVE 9216
__global__ __launch_bounds__(128) void k_attn(
    const unsigned short* __restrict__ QKVb,
    const unsigned short* __restrict__ Vt,
    const float* __restrict__ sinks,
    unsigned short* __restrict__ CTXb) {
  extern __shared__ unsigned short smem[];
  const int tid  = (int)threadIdx.x;
  const int wave = __builtin_amdgcn_readfirstlane(tid) >> 5;   // uniform (SGPR)
  const int lane = tid & 31, idx = lane & 15, half = lane >> 4;
  unsigned short* KB = smem + wave * (ATTN_LDS_PER_WAVE / 2);  // 32 keys x 64 ch
  unsigned short* VB = KB + 2048;                              // 64 d x 32 keys
  unsigned short* PB = VB + 2048;                              // 16 q x 32 keys
  const unsigned ldsK = (unsigned)(wave * ATTN_LDS_PER_WAVE);
  const unsigned ldsV = ldsK + 4096u;

  const int qt = blockIdx.x * 4 + wave;
  const int h = blockIdx.y, b = blockIdx.z, g = h >> 2;        // GS = 4
  const int i0 = qt * 16;
  const int t0 = b * SEQ + i0;

  // Q fragments for the two 32-channel halves of HD=64
  const unsigned short* qrow = QKVb + (size_t)(t0 + idx) * NQKV + h * HD;
  BFrag aQ0, aQ1;
  load_frag(aQ0, qrow, 0, half);
  load_frag(aQ1, qrow, 32, half);

  float m[8], l[8];
  const float sink = sinks[h];
  for (int r = 0; r < 8; ++r) { m[r] = sink; l[r] = 1.0f; }    // sink in init state
  v8f accO[4];
  for (int c = 0; c < 4; ++c) accO[c] = (v8f){};

  int jlo = i0 - (WIN - 1); if (jlo < 0) jlo = 0;
  const int kp0 = jlo & ~31;          // 32-aligned => kp+31 <= SEQ-1 always
  for (int kp = kp0; kp <= i0 + 15; kp += 32) {
#if USE_TDM
    // DMA the K tile (32 keys x 64 ch, row stride NQKV) and the V^T tile
    // (64 d x 32 keys, row stride SEQ) into LDS via the Tensor Data Mover.
    tdm_load_2d(ldsK, QKVb + (size_t)(b * SEQ + kp) * NQKV + NH * HD + g * HD,
                64u, 32u, (unsigned long long)NQKV);
    tdm_load_2d(ldsV, Vt + (((size_t)((b * NKV + g) * HD)) << 11) + kp,
                32u, 64u, (unsigned long long)SEQ);
    __builtin_amdgcn_s_wait_tensorcnt((unsigned short)0);
#endif
    v8f S[2];
    for (int td = 0; td < 2; ++td) {
#if USE_TDM
      const unsigned short* krow = KB + (td * 16 + idx) * 64;
#else
      const unsigned short* krow =
          QKVb + (size_t)(b * SEQ + kp + td * 16 + idx) * NQKV + NH * HD + g * HD;
#endif
      BFrag bK0, bK1;
      load_frag(bK0, krow, 0, half);
      load_frag(bK1, krow, 32, half);
      v8f sc = {};
      sc = wmma_bf16(aQ0, bK0, sc);
      sc = wmma_bf16(aQ1, bK1, sc);
      S[td] = sc;
    }
    // mask + online softmax over the 32 keys of this step
    float p0[8], p1[8];
    for (int r = 0; r < 8; ++r) {
      int i  = i0 + half * 8 + r;
      int ja = kp + idx, jb = kp + 16 + idx;
      float s0 = ((ja <= i) && (ja > i - WIN)) ? S[0][r] * 0.125f : -1e30f;
      float s1 = ((jb <= i) && (jb > i - WIN)) ? S[1][r] * 0.125f : -1e30f;
      float tm = fmaxf(s0, s1);
      for (int off = 1; off < 16; off <<= 1) tm = fmaxf(tm, __shfl_xor(tm, off, 32));
      float mn = fmaxf(m[r], tm);
      float scale = __expf(m[r] - mn);
      m[r] = mn; l[r] *= scale;
      for (int c = 0; c < 4; ++c) accO[c][r] *= scale;
      float e0 = __expf(s0 - mn), e1 = __expf(s1 - mn);
      float ps = e0 + e1;
      for (int off = 1; off < 16; off <<= 1) ps += __shfl_xor(ps, off, 32);
      l[r] += ps;
      p0[r] = e0; p1[r] = e1;
    }
    // stage P (lane=key layout -> LDS row-major) then reload as A-fragment
    for (int r = 0; r < 8; ++r) {
      PB[(half * 8 + r) * 32 + idx]      = f2bf(p0[r]);
      PB[(half * 8 + r) * 32 + 16 + idx] = f2bf(p1[r]);
    }
    __builtin_amdgcn_wave_barrier();
    BFrag aP;
    load_frag(aP, PB + idx * 32, 0, half);
    for (int c = 0; c < 4; ++c) {
#if USE_TDM
      const unsigned short* vrow = VB + (c * 16 + idx) * 32;
#else
      const unsigned short* vrow =
          Vt + (((size_t)((b * NKV + g) * HD + c * 16 + idx)) << 11) + kp;
#endif
      BFrag bV;
      load_frag(bV, vrow, 0, half);
      accO[c] = wmma_bf16(aP, bV, accO[c]);
    }
    __builtin_amdgcn_wave_barrier();
  }
  for (int c = 0; c < 4; ++c)
    for (int r = 0; r < 8; ++r) {
      float o = accO[c][r] / l[r];
      int row = t0 + half * 8 + r;
      CTXb[(size_t)row * EMB + h * HD + c * 16 + idx] = f2bf(o);
    }
}

extern "C" void kernel_launch(void* const* d_in, const int* in_sizes, int n_in,
                              void* d_out, int out_size, void* d_ws, size_t ws_size,
                              hipStream_t stream) {
  const float* x     = (const float*)d_in[0];
  const float* Wq    = (const float*)d_in[1];
  const float* bq    = (const float*)d_in[2];
  const float* Wk    = (const float*)d_in[3];
  const float* bk    = (const float*)d_in[4];
  const float* Wv    = (const float*)d_in[5];
  const float* bv    = (const float*)d_in[6];
  const float* Wo    = (const float*)d_in[7];
  const float* bo    = (const float*)d_in[8];
  const float* sinks = (const float*)d_in[9];
  const int*   pos   = (const int*)d_in[10];

  char* ws = (char*)d_ws;
  size_t off = 0;
  auto alloc = [&](size_t bytes) -> void* {
    void* p = ws + off;
    off += (bytes + 255) & ~(size_t)255;
    return p;
  };
  unsigned short* Xb    = (unsigned short*)alloc((size_t)TOK * EMB * 2);
  unsigned short* Wqkvb = (unsigned short*)alloc((size_t)NQKV * EMB * 2);
  unsigned short* Wob   = (unsigned short*)alloc((size_t)EMB * EMB * 2);
  unsigned short* QKVb  = (unsigned short*)alloc((size_t)TOK * NQKV * 2);
  unsigned short* Vtb   = (unsigned short*)alloc((size_t)TOK * NKV * HD * 2);
  unsigned short* CTXb  = (unsigned short*)alloc((size_t)TOK * EMB * 2);
  float*          biasq = (float*)alloc((size_t)NQKV * 4);

  // pack fp32 -> bf16 (weights row-concatenated: Q rows | K rows | V rows)
  k_f32_to_bf16<<<4096, 256, 0, stream>>>(x, Xb, TOK * EMB);
  k_f32_to_bf16<<<2048, 256, 0, stream>>>(Wq, Wqkvb, NH * HD * EMB);
  k_f32_to_bf16<<<512, 256, 0, stream>>>(Wk, Wqkvb + (size_t)NH * HD * EMB,
                                         NKV * HD * EMB);
  k_f32_to_bf16<<<512, 256, 0, stream>>>(
      Wv, Wqkvb + (size_t)(NH * HD + NKV * HD) * EMB, NKV * HD * EMB);
  k_f32_to_bf16<<<2048, 256, 0, stream>>>(Wo, Wob, EMB * EMB);
  hipMemcpyAsync(biasq, bq, NH * HD * 4, hipMemcpyDeviceToDevice, stream);
  hipMemcpyAsync(biasq + NH * HD, bk, NKV * HD * 4, hipMemcpyDeviceToDevice, stream);
  hipMemcpyAsync(biasq + NH * HD + NKV * HD, bv, NKV * HD * 4,
                 hipMemcpyDeviceToDevice, stream);

  // fused QKV projection (bf16 WMMA, fp32 accumulate, 64x64 tile per wave)
  k_gemm_qkv<<<dim3(NQKV / 256, TOK / 64), 128, 0, stream>>>(Xb, Wqkvb, biasq, QKVb);
  // YaRN RoPE on Q and K
  k_rope<<<(TOK * (NH + NKV) * (HD / 2) + 255) / 256, 256, 0, stream>>>(QKVb, pos);
  // V transpose for contiguous P·V B-fragments
  k_transpose_v<<<(TOK * NKV * HD + 255) / 256, 256, 0, stream>>>(QKVb, Vtb);
  // sliding-window attention with sink (TDM-staged K/V tiles when available)
  k_attn<<<dim3(SEQ / 16 / 4, NH, BATCH), 128, 4 * ATTN_LDS_PER_WAVE, stream>>>(
      QKVb, Vtb, sinks, CTXb);
  // output projection to fp32
  k_gemm_out<<<dim3(EMB / 256, TOK / 64), 128, 0, stream>>>(CTXb, Wob, bo,
                                                            (float*)d_out);
}
